// DeepFM_83769042141642
// MI455X (gfx1250) — compile-verified, matrix operand order checked by reading, not run on hardware
//
#include <hip/hip_runtime.h>
#include <hip/hip_bf16.h>

// ---------------------------------------------------------------------------
// DeepFM forward for MI455X (gfx1250, wave32, WMMA, async global->LDS).
//   K1: gather cat/num embeddings -> x16 (f16, B x 1248); fused lin + FM -> linfm
//   K2: f32 -> f16 weight conversion (W1, W2)
//   K3: WMMA GEMM + bias + ReLU, double-buffered LDS filled with
//       global_load_async_to_lds_b128 (ASYNCcnt), 8 x v_wmma per K-step
//   K4: head: out[b] = linfm[b] + dot(h2[b,:], Wo) + bo
// ---------------------------------------------------------------------------

#define NUM_CAT 26
#define CARD    100000
#define NUM_NUM 13
#define EMBED   32
#define H1      512
#define H2      256
#define BATCH   16384
#define DEEP_IN 1248   // 32 * 39

typedef __attribute__((ext_vector_type(16))) _Float16 v16h;
typedef __attribute__((ext_vector_type(8)))  _Float16 v8h;
typedef __attribute__((ext_vector_type(8)))  float    v8f;

union Frag16 { v16h f; v8h h[2]; };

// generic -> LDS(AS3) byte offset; runtime instruction (ptrtoint of AS3 ptr).
typedef __attribute__((address_space(3))) _Float16 lds_f16_t;
__device__ __forceinline__ uint32_t lds_off_u32(_Float16* p) {
    return (uint32_t)(uintptr_t)(lds_f16_t*)p;
}

// CDNA5 async DMA: global -> LDS, 16 bytes per lane, tracked by ASYNCcnt.
__device__ __forceinline__ void async_ld_b128(uint32_t lds_byte_off, const _Float16* g) {
    asm volatile("global_load_async_to_lds_b128 %0, %1, off"
                 :: "v"(lds_byte_off), "v"((uint64_t)(uintptr_t)g)
                 : "memory");
}
template<int N>
__device__ __forceinline__ void wait_asynccnt() {
    asm volatile("s_wait_asynccnt %0" :: "i"(N) : "memory");
}
__device__ __forceinline__ void wait_dscnt0() {
    asm volatile("s_wait_dscnt 0x0" ::: "memory");
}

// ---------------------------------------------------------------------------
// K1: embedding gather + FM(2nd order) + linear term, one wave per batch row,
//     lane = embedding dim (0..31).
// ---------------------------------------------------------------------------
__global__ __launch_bounds__(256)
void gather_fm_lin_kernel(const int* __restrict__ cat_x,        // (B, 26)
                          const float* __restrict__ num_x,      // (B, 13)
                          const float* __restrict__ lin_cat,    // (26, CARD)
                          const float* __restrict__ lin_num_w,  // (13)
                          const float* __restrict__ lin_bias,   // (1)
                          const float* __restrict__ fm_cat,     // (26, CARD, 32)
                          const float* __restrict__ fm_num,     // (13, 32)
                          _Float16* __restrict__ x16,           // (B, 1248)
                          float* __restrict__ linfm)            // (B)
{
    const int wid  = threadIdx.x >> 5;                  // wave in block (0..7)
    const int lane = threadIdx.x & 31;                  // embedding dim e
    const int b    = blockIdx.x * 8 + wid;
    if (b >= BATCH) return;

    const int* my_cat = cat_x + (size_t)b * NUM_CAT;
    _Float16*  my_x   = x16   + (size_t)b * DEEP_IN;

    float s = 0.f, sq = 0.f;

    // categorical fields: value == embedding
    #pragma unroll
    for (int f = 0; f < NUM_CAT; ++f) {
        const int id = my_cat[f];
        const float v = fm_cat[((size_t)f * CARD + (size_t)id) * EMBED + lane];
        my_x[f * EMBED + lane] = (_Float16)v;
        s  += v;
        sq += v * v;
    }
    // numerical fields: value == embedding * num_x
    #pragma unroll
    for (int j = 0; j < NUM_NUM; ++j) {
        const float e = fm_num[j * EMBED + lane];
        my_x[(NUM_CAT + j) * EMBED + lane] = (_Float16)e;
        const float v = e * num_x[(size_t)b * NUM_NUM + j];
        s  += v;
        sq += v * v;
    }

    // per-lane FM contribution: s_e^2 - sum_f v_e^2
    float red = 0.5f * (s * s - sq);

    // linear term folded into the same 32-lane reduction
    if (lane < NUM_CAT)
        red += lin_cat[(size_t)lane * CARD + (size_t)my_cat[lane]];
    if (lane < NUM_NUM)
        red += num_x[(size_t)b * NUM_NUM + lane] * lin_num_w[lane];

    // wave32 butterfly reduction
    #pragma unroll
    for (int off = 16; off > 0; off >>= 1)
        red += __shfl_xor(red, off, 32);

    if (lane == 0) linfm[b] = red + lin_bias[0];
}

// ---------------------------------------------------------------------------
// K2: f32 -> f16 conversion (weights)
// ---------------------------------------------------------------------------
__global__ __launch_bounds__(256)
void cvt_f32_to_f16_kernel(const float* __restrict__ src, _Float16* __restrict__ dst, int n)
{
    const int i = blockIdx.x * 256 + threadIdx.x;
    if (i < n) dst[i] = (_Float16)src[i];
}

// ---------------------------------------------------------------------------
// K3: WMMA GEMM:  C[m,n] = act( sum_k A[m,k] * Bw[n,k] + bias[n] )
//     A: (M,K) f16 row-major, Bw: (N,K) f16 row-major (weights), C: (M,N) f16.
//     Block = 128x128 C-tile, 8 waves (4 along M x 2 along N), wave = 32x64.
//     Double-buffered LDS tiles filled via async global->LDS DMA.
//     All LDS addressing is integer half-offsets into one `smem` array so no
//     generic pointers to LDS end up in static initializers.
// ---------------------------------------------------------------------------
#define BM   128
#define BN   128
#define BK   32
#define LDT  40                 // padded LDS row stride (halfs): 80B, 16B-aligned
#define TILE_HALFS (BM * LDT)   // one A (or B) tile
#define BUF_HALFS  (2 * TILE_HALFS)   // A tile + B tile per buffer

__global__ __launch_bounds__(256)
void wmma_gemm_bias_act_kernel(const _Float16* __restrict__ A,
                               const _Float16* __restrict__ Bw,
                               const float* __restrict__ bias,
                               _Float16* __restrict__ C,
                               int M, int N, int K, int relu)
{
    // layout (halfs): [buf0: A tile | B tile][buf1: A tile | B tile]
    __shared__ _Float16 smem[2 * BUF_HALFS];

    const int tid   = threadIdx.x;
    const int wid   = tid >> 5;
    const int lane  = tid & 31;
    const int lhalf = lane & 15;          // 0..15
    const int lhi   = lane >> 4;          // 0 or 1
    const int wm    = (wid & 3) * 32;     // wave M offset in tile
    const int wn    = (wid >> 2) * 64;    // wave N offset in tile
    const int bM    = blockIdx.x * BM;
    const int bN    = blockIdx.y * BN;

    // LDS byte offset of smem base (runtime ptrtoint, not a static init)
    const uint32_t lds_base = lds_off_u32(&smem[0]);

    // This thread's staging assignment: rows r0 and r0+64, 8-half column kc.
    const int r0 = tid >> 2;              // 0..63
    const int kc = (tid & 3) * 8;         // 0,8,16,24 (halfs)

    // Half-offsets (within one buffer) of the 4 chunks this thread fills.
    const uint32_t hA0 = (uint32_t)((r0)      * LDT + kc);
    const uint32_t hA1 = (uint32_t)((r0 + 64) * LDT + kc);
    const uint32_t hB0 = (uint32_t)(TILE_HALFS + (r0)      * LDT + kc);
    const uint32_t hB1 = (uint32_t)(TILE_HALFS + (r0 + 64) * LDT + kc);

    const _Float16* gA0 = A  + (size_t)(bM + r0)      * K + kc;
    const _Float16* gA1 = A  + (size_t)(bM + r0 + 64) * K + kc;
    const _Float16* gB0 = Bw + (size_t)(bN + r0)      * K + kc;
    const _Float16* gB1 = Bw + (size_t)(bN + r0 + 64) * K + kc;

    // K-offset of the first contiguous 8-half group this lane owns:
    // lanes 0-15: K = 0..7 & 16..23 ; lanes 16-31: K = 8..15 & 24..31
    const int koff = lhi * 8;

    v8f acc[2][4];
    const v8f vzero = {0.f, 0.f, 0.f, 0.f, 0.f, 0.f, 0.f, 0.f};
    #pragma unroll
    for (int mi = 0; mi < 2; ++mi)
        #pragma unroll
        for (int ni = 0; ni < 4; ++ni)
            acc[mi][ni] = vzero;

    const int ntiles = K / BK;

    // prologue: async-fill tile 0 into buffer 0 (4 DMA ops per wave)
    async_ld_b128(lds_base + 2u * hA0, gA0);
    async_ld_b128(lds_base + 2u * hA1, gA1);
    async_ld_b128(lds_base + 2u * hB0, gB0);
    async_ld_b128(lds_base + 2u * hB1, gB1);

    int buf = 0;
    for (int t = 0; t < ntiles; ++t) {
        if (t + 1 < ntiles) {
            // async-fill next tile into the other buffer, then wait only for
            // the older 4 DMAs (this tile) -- async ops complete in order.
            const int knext = (t + 1) * BK;
            const uint32_t nb = lds_base + 2u * (uint32_t)((buf ^ 1) * BUF_HALFS);
            async_ld_b128(nb + 2u * hA0, gA0 + knext);
            async_ld_b128(nb + 2u * hA1, gA1 + knext);
            async_ld_b128(nb + 2u * hB0, gB0 + knext);
            async_ld_b128(nb + 2u * hB1, gB1 + knext);
            wait_asynccnt<4>();
        } else {
            wait_asynccnt<0>();
        }
        __syncthreads();   // all waves' fills for `buf` are visible

        // --- load fragments (two ds_load_b128 per fragment per lane) ---
        _Float16* curA = smem + buf * BUF_HALFS;
        _Float16* curB = curA + TILE_HALFS;

        Frag16 af[2], bf[4];
        #pragma unroll
        for (int mi = 0; mi < 2; ++mi) {
            _Float16* p = curA + (wm + mi * 16 + lhalf) * LDT + koff;
            af[mi].h[0] = *(const v8h*)(p);
            af[mi].h[1] = *(const v8h*)(p + 16);
        }
        #pragma unroll
        for (int ni = 0; ni < 4; ++ni) {
            _Float16* p = curB + (wn + ni * 16 + lhalf) * LDT + koff;
            bf[ni].h[0] = *(const v8h*)(p);
            bf[ni].h[1] = *(const v8h*)(p + 16);
        }

        // --- 8 WMMAs: 32x64 wave tile, K += 32 ---
        #pragma unroll
        for (int mi = 0; mi < 2; ++mi)
            #pragma unroll
            for (int ni = 0; ni < 4; ++ni)
                acc[mi][ni] = __builtin_amdgcn_wmma_f32_16x16x32_f16(
                    /*neg_a=*/false, af[mi].f,
                    /*neg_b=*/false, bf[ni].f,
                    /*c_mod=*/(short)0, acc[mi][ni],
                    /*reuse_a=*/false, /*reuse_b=*/false);

        // make sure this wave's LDS reads retired before anyone refills `buf`
        wait_dscnt0();
        __syncthreads();
        buf ^= 1;
    }

    // --- epilogue: bias + (optional) ReLU, store f16 ---
    // C/D layout: VGPR r, lanes 0-15 -> M=r, N=lane ; lanes 16-31 -> M=r+8, N=lane-16
    #pragma unroll
    for (int ni = 0; ni < 4; ++ni) {
        const int col = bN + wn + ni * 16 + lhalf;
        const float bv = bias[col];
        #pragma unroll
        for (int mi = 0; mi < 2; ++mi) {
            #pragma unroll
            for (int r = 0; r < 8; ++r) {
                const int row = bM + wm + mi * 16 + lhi * 8 + r;
                float v = acc[mi][ni][r] + bv;
                if (relu) v = v > 0.f ? v : 0.f;
                C[(size_t)row * N + col] = (_Float16)v;
            }
        }
    }
}

// ---------------------------------------------------------------------------
// K4: head: out[b] = linfm[b] + dot(h2[b, 0:256], Wo) + bo   (wave per row)
// ---------------------------------------------------------------------------
__global__ __launch_bounds__(256)
void head_kernel(const _Float16* __restrict__ h2,    // (B, 256)
                 const float* __restrict__ Wo,       // (256)
                 const float* __restrict__ bo,       // (1)
                 const float* __restrict__ linfm,    // (B)
                 float* __restrict__ out)            // (B)
{
    const int wid  = threadIdx.x >> 5;
    const int lane = threadIdx.x & 31;
    const int b    = blockIdx.x * 8 + wid;
    if (b >= BATCH) return;

    const _Float16* row = h2 + (size_t)b * H2;
    float acc = 0.f;
    #pragma unroll
    for (int i = 0; i < H2 / 32; ++i) {
        const int k = lane + i * 32;
        acc += (float)row[k] * Wo[k];
    }
    #pragma unroll
    for (int off = 16; off > 0; off >>= 1)
        acc += __shfl_xor(acc, off, 32);

    if (lane == 0) out[b] = linfm[b] + acc + bo[0];
}

// ---------------------------------------------------------------------------
// Launcher
// ---------------------------------------------------------------------------
extern "C" void kernel_launch(void* const* d_in, const int* in_sizes, int n_in,
                              void* d_out, int out_size, void* d_ws, size_t ws_size,
                              hipStream_t stream)
{
    const int*   cat_x     = (const int*)  d_in[0];
    const float* num_x     = (const float*)d_in[1];
    const float* lin_cat   = (const float*)d_in[2];
    const float* lin_num_w = (const float*)d_in[3];
    const float* lin_bias  = (const float*)d_in[4];
    const float* fm_cat    = (const float*)d_in[5];
    const float* fm_num    = (const float*)d_in[6];
    const float* W1        = (const float*)d_in[7];
    const float* b1        = (const float*)d_in[8];
    const float* W2        = (const float*)d_in[9];
    const float* b2        = (const float*)d_in[10];
    const float* Wo        = (const float*)d_in[11];
    const float* bo        = (const float*)d_in[12];
    float*       out       = (float*)d_out;

    // workspace layout (all offsets 256B-aligned)
    char* ws = (char*)d_ws;
    size_t off = 0;
    auto take = [&](size_t bytes) -> char* {
        char* p = ws + off;
        off = (off + bytes + 255) & ~(size_t)255;
        return p;
    };
    _Float16* x16   = (_Float16*)take((size_t)BATCH * DEEP_IN * 2);  // 40.9 MB
    _Float16* h1    = (_Float16*)take((size_t)BATCH * H1 * 2);       // 16.8 MB
    _Float16* h2    = (_Float16*)take((size_t)BATCH * H2 * 2);       //  8.4 MB
    _Float16* w1h   = (_Float16*)take((size_t)H1 * DEEP_IN * 2);     //  1.3 MB
    _Float16* w2h   = (_Float16*)take((size_t)H2 * H1 * 2);          //  0.3 MB
    float*    linfm = (float*)   take((size_t)BATCH * 4);            //  64 KB

    // K1: gather + lin + FM  (8 rows per 256-thread block)
    gather_fm_lin_kernel<<<BATCH / 8, 256, 0, stream>>>(
        cat_x, num_x, lin_cat, lin_num_w, lin_bias, fm_cat, fm_num, x16, linfm);

    // K2: weight conversion
    {
        const int n1 = H1 * DEEP_IN;
        cvt_f32_to_f16_kernel<<<(n1 + 255) / 256, 256, 0, stream>>>(W1, w1h, n1);
        const int n2 = H2 * H1;
        cvt_f32_to_f16_kernel<<<(n2 + 255) / 256, 256, 0, stream>>>(W2, w2h, n2);
    }

    // K3a: h1 = relu(x @ W1^T + b1)   M=16384, N=512, K=1248
    {
        dim3 grid(BATCH / BM, H1 / BN);
        wmma_gemm_bias_act_kernel<<<grid, 256, 0, stream>>>(
            x16, w1h, b1, h1, BATCH, H1, DEEP_IN, 1);
    }
    // K3b: h2 = relu(h1 @ W2^T + b2)  M=16384, N=256, K=512
    {
        dim3 grid(BATCH / BM, H2 / BN);
        wmma_gemm_bias_act_kernel<<<grid, 256, 0, stream>>>(
            h1, w2h, b2, h2, BATCH, H2, H1, 1);
    }

    // K4: head
    head_kernel<<<BATCH / 8, 256, 0, stream>>>(h2, Wo, bo, linfm, out);

    (void)in_sizes; (void)n_in; (void)out_size; (void)ws_size;
}
